// Attention_72516227825845
// MI455X (gfx1250) — compile-verified, matrix operand order checked by reading, not run on hardware
//
#include <hip/hip_runtime.h>
#include <hip/hip_bf16.h>

typedef __attribute__((ext_vector_type(16))) __bf16 v16bf;
typedef __attribute__((ext_vector_type(8)))  float  v8f;

#define NUM_HEADS 12
#define NSEQ      256
#define HD        64
#define CDIM      768
#define VSTR      68   // padded k-major stride for v tile in LDS

static __device__ __forceinline__ v8f zero8() {
    v8f z;
#pragma unroll
    for (int i = 0; i < 8; ++i) z[i] = 0.f;
    return z;
}

static __device__ __forceinline__ v8f wmma_bf16(v16bf a, v16bf b, v8f c) {
    return __builtin_amdgcn_wmma_f32_16x16x32_bf16(false, a, false, b, (short)0, c, false, false);
}

// A/B fragment (16x32 bf16) from a row-major LDS tile: element (row, k) at base[row*stride + k].
// ISA 7.12.2: lane L<16 -> M=L, K pairs; lanes 16..31 -> same M set, K offset +8.
// VGPR v holds K pair base (v%4)*2 + 16*(v/4).
static __device__ __forceinline__ v16bf frag_rowmajor_bf16(const __bf16* base, int row0, int kb,
                                                           int stride, int lane) {
    const int m  = row0 + (lane & 15);
    const int hi = (lane >> 4) & 1;
    const __bf16* p = base + m * stride + kb + 8 * hi;
    v16bf f;
#pragma unroll
    for (int v = 0; v < 8; ++v) {
        const int k = ((v & 3) << 1) + ((v >> 2) << 4);
        f[2 * v]     = p[k];
        f[2 * v + 1] = p[k + 1];
    }
    return f;
}

// Same layout, but source is f32 in LDS (attention matrix) -> convert to bf16 on the fly.
static __device__ __forceinline__ v16bf frag_rowmajor_f32(const float* base, int row0, int kb,
                                                          int stride, int lane) {
    const int m  = row0 + (lane & 15);
    const int hi = (lane >> 4) & 1;
    const float* p = base + m * stride + kb + 8 * hi;
    v16bf f;
#pragma unroll
    for (int v = 0; v < 8; ++v) {
        const int k = ((v & 3) << 1) + ((v >> 2) << 4);
        f[2 * v]     = (__bf16)p[k];
        f[2 * v + 1] = (__bf16)p[k + 1];
    }
    return f;
}

// B fragment (32x16 bf16) from a K-major LDS tile: element (k, n) at base[k*stride + n].
static __device__ __forceinline__ v16bf frag_kmajor_bf16(const __bf16* base, int kb, int n0,
                                                         int stride, int lane) {
    const int n  = n0 + (lane & 15);
    const int hi = (lane >> 4) & 1;
    v16bf f;
#pragma unroll
    for (int v = 0; v < 8; ++v) {
        const int k = kb + ((v & 3) << 1) + ((v >> 2) << 4) + 8 * hi;
        f[2 * v]     = base[k * stride + n];
        f[2 * v + 1] = base[(k + 1) * stride + n];
    }
    return f;
}

// ---------------------------------------------------------------------------
// Kernel 1: QV GEMM.  Y[m][n] = sum_k X[m][k] * Wqkv[row(n)][k], n in [0,1536).
// n<768 -> q third (w rows 0..767); n>=768 -> v third (w rows 1536..2303). k third skipped.
// Output bf16 [16384][1536].
// ---------------------------------------------------------------------------
__global__ __launch_bounds__(256) void qv_gemm_kernel(const float* __restrict__ X,
                                                      const float* __restrict__ Wqkv,
                                                      __bf16* __restrict__ QV) {
    __shared__ __bf16 As[128 * 34];
    __shared__ __bf16 Bs[128 * 34];
    const int tid  = threadIdx.x;
    const int lane = tid & 31;
    const int w    = tid >> 5;
    const int wr   = w & 3;   // tile-row pair
    const int wc   = w >> 2;  // tile-col quad
    const int m0   = blockIdx.x * 128;
    const int n0   = blockIdx.y * 128;

    v8f acc[2][4];
#pragma unroll
    for (int i = 0; i < 2; ++i)
#pragma unroll
        for (int j = 0; j < 4; ++j) acc[i][j] = zero8();

    for (int kb = 0; kb < CDIM; kb += 32) {
        __syncthreads();
#pragma unroll
        for (int i = 0; i < 4; ++i) {
            const int r = (tid >> 3) + 32 * i;
            const int q = (tid & 7) * 4;
            const float4 va = *reinterpret_cast<const float4*>(X + (size_t)(m0 + r) * CDIM + kb + q);
            __bf16* da = As + r * 34 + q;
            da[0] = (__bf16)va.x; da[1] = (__bf16)va.y; da[2] = (__bf16)va.z; da[3] = (__bf16)va.w;
            const int n    = n0 + r;
            const int wrow = n + (n >= 768 ? 768 : 0);
            const float4 vb = *reinterpret_cast<const float4*>(Wqkv + (size_t)wrow * CDIM + kb + q);
            __bf16* db = Bs + r * 34 + q;
            db[0] = (__bf16)vb.x; db[1] = (__bf16)vb.y; db[2] = (__bf16)vb.z; db[3] = (__bf16)vb.w;
        }
        __syncthreads();
        const v16bf a0 = frag_rowmajor_bf16(As, 16 * (2 * wr), 0, 34, lane);
        const v16bf a1 = frag_rowmajor_bf16(As, 16 * (2 * wr + 1), 0, 34, lane);
#pragma unroll
        for (int j = 0; j < 4; ++j) {
            const v16bf b = frag_rowmajor_bf16(Bs, 16 * (4 * wc + j), 0, 34, lane);
            acc[0][j] = wmma_bf16(a0, b, acc[0][j]);
            acc[1][j] = wmma_bf16(a1, b, acc[1][j]);
        }
    }

    const int hi = (lane >> 4) & 1;
    const int nn = lane & 15;
#pragma unroll
    for (int i = 0; i < 2; ++i)
#pragma unroll
        for (int j = 0; j < 4; ++j)
#pragma unroll
            for (int r = 0; r < 8; ++r) {
                const int gm = m0 + 16 * (2 * wr + i) + r + 8 * hi;
                const int gn = n0 + 16 * (4 * wc + j) + nn;
                QV[(size_t)gm * 1536 + gn] = (__bf16)acc[i][j][r];
            }
}

// ---------------------------------------------------------------------------
// Kernel 2: per-(b,h) self-OT attention.  One workgroup per head instance.
// LDS: logK f32 [256][256] (256KB) + bf16 tile (qn then v, 34KB) + logu/logv.
// ---------------------------------------------------------------------------
__global__ __launch_bounds__(256) void sot_attn_kernel(const __bf16* __restrict__ QV,
                                                       __bf16* __restrict__ OutBH) {
    extern __shared__ char smem_raw[];
    float*  logK = reinterpret_cast<float*>(smem_raw);
    __bf16* tile = reinterpret_cast<__bf16*>(smem_raw + NSEQ * NSEQ * 4);
    float*  logu = reinterpret_cast<float*>(smem_raw + NSEQ * NSEQ * 4 + NSEQ * VSTR * 2);
    float*  logv = logu + NSEQ;

    const int tid  = threadIdx.x;
    const int lane = tid & 31;
    const int w    = tid >> 5;
    const int bh   = blockIdx.x;
    const int b    = bh / NUM_HEADS;
    const int h    = bh % NUM_HEADS;

    // Phase 1: load q row, normalize (q / (||q|| + 1e-8)), store bf16 qn to LDS (stride HD).
    {
        const int n = tid;
        const __bf16* qg = QV + (size_t)(b * NSEQ + n) * 1536 + h * HD;
        float ss = 0.f;
#pragma unroll
        for (int d = 0; d < HD; ++d) { const float q = (float)qg[d]; ss += q * q; }
        const float rn = 1.0f / (sqrtf(ss) + 1e-8f);
#pragma unroll
        for (int d = 0; d < HD; ++d) tile[n * HD + d] = (__bf16)((float)qg[d] * rn);
        logv[n] = 0.f;
    }
    __syncthreads();

    // Phase 2: gram via WMMA -> log_K = -(1 - s)/0.1 = 10*(s-1); diag = -DIAG_VAL/0.1.
    for (int t = w; t < 256; t += 8) {
        const int ti = t >> 4;
        const int tj = t & 15;
        v8f acc = zero8();
        acc = wmma_bf16(frag_rowmajor_bf16(tile, 16 * ti, 0, HD, lane),
                        frag_rowmajor_bf16(tile, 16 * tj, 0, HD, lane), acc);
        acc = wmma_bf16(frag_rowmajor_bf16(tile, 16 * ti, 32, HD, lane),
                        frag_rowmajor_bf16(tile, 16 * tj, 32, HD, lane), acc);
        const int nn = lane & 15;
        const int hi = (lane >> 4) & 1;
#pragma unroll
        for (int r = 0; r < 8; ++r) {
            const int M = 16 * ti + r + 8 * hi;
            const int N = 16 * tj + nn;
            float lk = 10.0f * (acc[r] - 1.0f);
            if (M == N) lk = -10000.0f;
            logK[M * NSEQ + N] = lk;
        }
    }
    __syncthreads();

    // Phase 3: Sinkhorn (10 iterations of row/col logsumexp).
    const float log_marg = -5.545177444479562f;  // -log(256)
    for (int it = 0; it < 10; ++it) {
        {
            const int n = tid;
            const float* row = logK + n * NSEQ;
            float mx = -3.4e38f;
            for (int m = 0; m < NSEQ; ++m) mx = fmaxf(mx, row[m] + logv[m]);
            float s = 0.f;
            for (int m = 0; m < NSEQ; ++m) s += __expf(row[m] + logv[m] - mx);
            logu[n] = log_marg - (mx + __logf(s));
        }
        __syncthreads();
        {
            const int m = tid;
            float mx = -3.4e38f;
            for (int n = 0; n < NSEQ; ++n) mx = fmaxf(mx, logK[n * NSEQ + m] + logu[n]);
            float s = 0.f;
            for (int n = 0; n < NSEQ; ++n) s += __expf(logK[n * NSEQ + m] + logu[n] - mx);
            logv[m] = log_marg - (mx + __logf(s));
        }
        __syncthreads();
    }

    // Phase 4: z = exp(log_u + log_K + log_v) in place; per-row max into logu.
    {
        const int n  = tid;
        const float lu = logu[n];
        float* row = logK + n * NSEQ;
        float mx = 0.f;
        for (int m = 0; m < NSEQ; ++m) {
            const float z = __expf(lu + row[m] + logv[m]);
            row[m] = z;
            mx = fmaxf(mx, z);
        }
        logu[n] = mx;
    }
    __syncthreads();
    for (int s = 128; s > 0; s >>= 1) {
        if (tid < s) logu[tid] = fmaxf(logu[tid], logu[tid + s]);
        __syncthreads();
    }
    const float invg = 1.0f / logu[0];

    // Phase 5: z /= gmax; diag -> 1; softmax(z * 0.125) per row (row max is 0.125, the diagonal).
    {
        const int n = tid;
        float* row = logK + n * NSEQ;
        const float scale = 0.125f;
        float s = 0.f;
        for (int m = 0; m < NSEQ; ++m) {
            const float val = (m == n) ? scale : scale * row[m] * invg;
            const float e = __expf(val - scale);
            row[m] = e;
            s += e;
        }
        const float inv = 1.0f / s;
        for (int m = 0; m < NSEQ; ++m) row[m] *= inv;
    }
    __syncthreads();

    // Phase 6: load v (bf16) into tile, K-major [m][d] with padded stride VSTR.
    {
        const int n = tid;
        const __bf16* vg = QV + (size_t)(b * NSEQ + n) * 1536 + 768 + h * HD;
#pragma unroll
        for (int d = 0; d < HD; ++d) tile[n * VSTR + d] = vg[d];
    }
    __syncthreads();

    // Phase 7: out = attn @ v via WMMA. Wave w: output tile rows 2w, 2w+1, cols 0..3.
    const int nn = lane & 15;
    const int hi = (lane >> 4) & 1;
#pragma unroll
    for (int i = 0; i < 2; ++i) {
        const int ti = 2 * w + i;
#pragma unroll
        for (int j = 0; j < 4; ++j) {
            v8f acc = zero8();
#pragma unroll
            for (int kc = 0; kc < 8; ++kc) {
                const v16bf a  = frag_rowmajor_f32(logK, 16 * ti, 32 * kc, NSEQ, lane);
                const v16bf bv = frag_kmajor_bf16(tile, 32 * kc, 16 * j, VSTR, lane);
                acc = wmma_bf16(a, bv, acc);
            }
#pragma unroll
            for (int r = 0; r < 8; ++r) {
                const int gm = b * NSEQ + 16 * ti + r + 8 * hi;
                const int gc = h * HD + 16 * j + nn;
                OutBH[(size_t)gm * CDIM + gc] = (__bf16)acc[r];
            }
        }
    }
}

// ---------------------------------------------------------------------------
// Kernel 3: projection GEMM + bias.  Out[m][n] = sum_k A[m][k]*Wp[n][k] + bias[n], f32 out.
// ---------------------------------------------------------------------------
__global__ __launch_bounds__(256) void proj_gemm_kernel(const __bf16* __restrict__ A,
                                                        const float* __restrict__ Wp,
                                                        const float* __restrict__ bias,
                                                        float* __restrict__ Out) {
    __shared__ __bf16 As[128 * 34];
    __shared__ __bf16 Bs[128 * 34];
    const int tid  = threadIdx.x;
    const int lane = tid & 31;
    const int w    = tid >> 5;
    const int wr   = w & 3;
    const int wc   = w >> 2;
    const int m0   = blockIdx.x * 128;
    const int n0   = blockIdx.y * 128;

    v8f acc[2][4];
#pragma unroll
    for (int i = 0; i < 2; ++i)
#pragma unroll
        for (int j = 0; j < 4; ++j) acc[i][j] = zero8();

    for (int kb = 0; kb < CDIM; kb += 32) {
        __syncthreads();
#pragma unroll
        for (int i = 0; i < 4; ++i) {
            const int r = (tid >> 3) + 32 * i;
            const int q = (tid & 7) * 4;
            const __bf16* sa = A + (size_t)(m0 + r) * CDIM + kb + q;
            __bf16* da = As + r * 34 + q;
            da[0] = sa[0]; da[1] = sa[1]; da[2] = sa[2]; da[3] = sa[3];
            const int n = n0 + r;
            const float4 vb = *reinterpret_cast<const float4*>(Wp + (size_t)n * CDIM + kb + q);
            __bf16* db = Bs + r * 34 + q;
            db[0] = (__bf16)vb.x; db[1] = (__bf16)vb.y; db[2] = (__bf16)vb.z; db[3] = (__bf16)vb.w;
        }
        __syncthreads();
        const v16bf a0 = frag_rowmajor_bf16(As, 16 * (2 * wr), 0, 34, lane);
        const v16bf a1 = frag_rowmajor_bf16(As, 16 * (2 * wr + 1), 0, 34, lane);
#pragma unroll
        for (int j = 0; j < 4; ++j) {
            const v16bf b = frag_rowmajor_bf16(Bs, 16 * (4 * wc + j), 0, 34, lane);
            acc[0][j] = wmma_bf16(a0, b, acc[0][j]);
            acc[1][j] = wmma_bf16(a1, b, acc[1][j]);
        }
    }

    const int hi = (lane >> 4) & 1;
    const int nn = lane & 15;
#pragma unroll
    for (int i = 0; i < 2; ++i)
#pragma unroll
        for (int j = 0; j < 4; ++j)
#pragma unroll
            for (int r = 0; r < 8; ++r) {
                const int gm = m0 + 16 * (2 * wr + i) + r + 8 * hi;
                const int gn = n0 + 16 * (4 * wc + j) + nn;
                Out[(size_t)gm * CDIM + gn] = acc[i][j][r] + bias[gn];
            }
}

extern "C" void kernel_launch(void* const* d_in, const int* in_sizes, int n_in,
                              void* d_out, int out_size, void* d_ws, size_t ws_size,
                              hipStream_t stream) {
    (void)in_sizes; (void)n_in; (void)out_size; (void)ws_size;
    const float* x      = (const float*)d_in[0];
    const float* w_qkv  = (const float*)d_in[1];
    const float* w_proj = (const float*)d_in[2];
    const float* b_proj = (const float*)d_in[3];
    float* out = (float*)d_out;

    const size_t M = 64 * 256;  // 16384 rows
    __bf16* qv  = (__bf16*)d_ws;                                   // [16384][1536] bf16
    __bf16* obh = (__bf16*)((char*)d_ws + M * 1536 * sizeof(__bf16)); // [16384][768] bf16

    qv_gemm_kernel<<<dim3(128, 12), 256, 0, stream>>>(x, w_qkv, qv);

    const size_t shmem = (size_t)NSEQ * NSEQ * 4 + (size_t)NSEQ * VSTR * 2 + 2 * NSEQ * 4;
    sot_attn_kernel<<<dim3(64 * NUM_HEADS), 256, shmem, stream>>>(qv, obh);

    proj_gemm_kernel<<<dim3(128, 6), 256, 0, stream>>>(obh, w_proj, b_proj, out);
}
